// GCNEncoder_48215302865186
// MI455X (gfx1250) — matrix operand order repaired and checked by reading, not compiled, not run on hardware
//
#include <hip/hip_runtime.h>
#include <math.h>

#define Bn 8
#define Vn 200
#define En 40000
#define Hn 128
#define Ln 3
#define TILE_M 128
#define SW_STRIDE 136   // padded f16 stride: lane step 272B -> bank step 4 -> conflict-free 16B reads

typedef __attribute__((ext_vector_type(16))) _Float16 v16h;
typedef __attribute__((ext_vector_type(8)))  _Float16 v8h;
typedef __attribute__((ext_vector_type(8)))  float    v8f;

__device__ __forceinline__ float sigmoidf_(float x) { return 1.0f / (1.0f + __expf(-x)); }

// ---------------- embed ----------------
__global__ void k_embed_nodes(const float* __restrict__ x, const float* __restrict__ nW,
                              const float* __restrict__ nb, float* __restrict__ xh) {
    int idx = blockIdx.x * blockDim.x + threadIdx.x;
    if (idx >= Bn * Vn * Hn) return;
    int h = idx % Hn;
    int bv = idx / Hn;
    xh[idx] = x[bv * 2 + 0] * nW[h] + x[bv * 2 + 1] * nW[Hn + h] + nb[h];
}

__global__ void k_embed_edges(const float* __restrict__ e, const float* __restrict__ eW,
                              const float* __restrict__ eb, float* __restrict__ eh) {
    size_t total = (size_t)Bn * En * Hn;
    for (size_t idx = blockIdx.x * (size_t)blockDim.x + threadIdx.x; idx < total;
         idx += (size_t)gridDim.x * blockDim.x) {
        int h = (int)(idx % Hn);
        size_t be = idx / Hn;
        eh[idx] = e[be] * eW[h] + eb[h];
    }
}

__global__ void k_zero(float* __restrict__ p, int n) {
    int i = blockIdx.x * blockDim.x + threadIdx.x;
    if (i < n) p[i] = 0.f;
}

// ---------------- node-side small GEMMs: out = xh @ W + b (4 weight sets) ----------------
__global__ void k_node(const float* __restrict__ xh,
                       const float* __restrict__ Uw, const float* __restrict__ Ub,
                       const float* __restrict__ Vw, const float* __restrict__ Vb_,
                       const float* __restrict__ Bw, const float* __restrict__ Bb,
                       const float* __restrict__ Cw, const float* __restrict__ Cb,
                       float* __restrict__ Ux, float* __restrict__ Vx,
                       float* __restrict__ Bx, float* __restrict__ Cx) {
    int b = blockIdx.y;
    int idx = blockIdx.x * blockDim.x + threadIdx.x;
    if (idx >= 4 * Vn * Hn) return;
    int mat = idx / (Vn * Hn);
    int rem = idx % (Vn * Hn);
    int v = rem / Hn, h = rem % Hn;
    const float *W, *bias;
    float* out;
    switch (mat) {
        case 0:  W = Uw; bias = Ub;  out = Ux; break;
        case 1:  W = Vw; bias = Vb_; out = Vx; break;
        case 2:  W = Bw; bias = Bb;  out = Bx; break;
        default: W = Cw; bias = Cb;  out = Cx; break;
    }
    const float* xr = xh + ((size_t)b * Vn + v) * Hn;
    float acc = bias[h];
    for (int k = 0; k < Hn; ++k) acc += xr[k] * W[k * Hn + h];
    out[((size_t)b * Vn + v) * Hn + h] = acc;
}

// ---------------- fused WMMA edge kernel ----------------
// mode 0: eh += relu(eh @ W + bias + Bx[src] + Cx[dst]); agg[dst] += sigmoid(eh_pre) * Vx[src]
// mode 1: eh  = relu(eh @ W + bias)           (final out_W1 layer; h overwrites eh)
__global__ __launch_bounds__(256) void k_edge(
    float* __restrict__ eh, const float* __restrict__ W, const float* __restrict__ bias,
    const float* __restrict__ Bx, const float* __restrict__ Cx,
    const float* __restrict__ Vx, float* __restrict__ agg,
    const int* __restrict__ eidx, int mode) {
    __shared__ _Float16 sW[128 * SW_STRIDE];     // W transposed: sW[n*S + k]
    __shared__ _Float16 sE[TILE_M * SW_STRIDE];  // eh tile:      sE[m*S + k]
    __shared__ int sSrc[TILE_M];
    __shared__ int sDst[TILE_M];

    const int b    = blockIdx.y;
    const int base = blockIdx.x * TILE_M;
    const int tid  = threadIdx.x;
    const int lane = tid & 31;
    const int wid  = tid >> 5;

    // stage weight (coalesced global read, f32->f16, transposed into LDS)
    for (int i = tid; i < 128 * 128; i += 256) {
        int k = i >> 7, n = i & 127;
        sW[n * SW_STRIDE + k] = (_Float16)W[i];
    }
    // stage eh tile (zero-pad past E)
    for (int i = tid; i < TILE_M * 128; i += 256) {
        int el = i >> 7, h = i & 127;
        int edge = base + el;
        float v = (edge < En) ? eh[((size_t)b * En + edge) * Hn + h] : 0.f;
        sE[el * SW_STRIDE + h] = (_Float16)v;
    }
    if (mode == 0) {
        for (int i = tid; i < TILE_M; i += 256) {
            int edge = base + i;
            sSrc[i] = (edge < En) ? eidx[(size_t)b * 2 * En + edge] : 0;
            sDst[i] = (edge < En) ? eidx[(size_t)b * 2 * En + En + edge] : 0;
        }
    }
    __syncthreads();

    // each wave: 16 rows (edges) x 128 cols, K=128 in 4 steps of 32
    v8f acc[8] = {};
    const int m   = wid * 16 + (lane & 15);
    const int hiA = (lane >> 4) << 4;   // +16 K-offset for upper half-wave (A frag)
    const int hiB = (lane >> 4) << 4;   // +16 K-offset for upper half-wave (B frag)
    const int hiM = (lane >> 4) << 3;   // +8 row offset in C/D frag

#pragma unroll
    for (int ks = 0; ks < 4; ++ks) {
        v16h a;
        const _Float16* pe = &sE[m * SW_STRIDE + ks * 32 + hiA];
        ((v8h*)&a)[0] = *(const v8h*)(pe);        // K = kb .. kb+7
        ((v8h*)&a)[1] = *(const v8h*)(pe + 8);    // K = kb+8 .. kb+15
#pragma unroll
        for (int nt = 0; nt < 8; ++nt) {
            int n = nt * 16 + (lane & 15);
            const _Float16* pw = &sW[n * SW_STRIDE + ks * 32 + hiB];
            v16h bm;
            ((v8h*)&bm)[0] = *(const v8h*)(pw);       // K = kb .. kb+7
            ((v8h*)&bm)[1] = *(const v8h*)(pw + 8);   // K = kb+8 .. kb+15
            acc[nt] = __builtin_amdgcn_wmma_f32_16x16x32_f16(
                false, a, false, bm, (short)0, acc[nt], false, false);
        }
    }

    if (mode == 0) {
#pragma unroll
        for (int nt = 0; nt < 8; ++nt) {
            int n = nt * 16 + (lane & 15);
#pragma unroll
            for (int r = 0; r < 8; ++r) {
                int ml = wid * 16 + r + hiM;
                int edge = base + ml;
                if (edge < En) {
                    int s = sSrc[ml], d = sDst[ml];
                    float val = acc[nt][r] + bias[n]
                              + Bx[((size_t)b * Vn + s) * Hn + n]
                              + Cx[((size_t)b * Vn + d) * Hn + n];
                    size_t off = ((size_t)b * En + edge) * Hn + n;
                    eh[off] = eh[off] + fmaxf(val, 0.f);
                }
            }
        }
        // agg[dst] += sigmoid(eh_pre) * Vx[src]  (uses pre-update eh staged in LDS)
        for (int i = tid; i < TILE_M * 128; i += 256) {
            int el = i >> 7, h = i & 127;
            int edge = base + el;
            if (edge < En) {
                float g = sigmoidf_((float)sE[el * SW_STRIDE + h]);
                int s = sSrc[el], d = sDst[el];
                float contrib = g * Vx[((size_t)b * Vn + s) * Hn + h];
                atomicAdd(&agg[((size_t)b * Vn + d) * Hn + h], contrib);
            }
        }
    } else {
#pragma unroll
        for (int nt = 0; nt < 8; ++nt) {
            int n = nt * 16 + (lane & 15);
#pragma unroll
            for (int r = 0; r < 8; ++r) {
                int ml = wid * 16 + r + hiM;
                int edge = base + ml;
                if (edge < En)
                    eh[((size_t)b * En + edge) * Hn + n] = fmaxf(acc[nt][r] + bias[n], 0.f);
            }
        }
    }
}

// ---------------- x update: xh += relu(Ux + agg) ----------------
__global__ void k_xupd(float* __restrict__ xh, const float* __restrict__ Ux,
                       const float* __restrict__ agg) {
    int i = blockIdx.x * blockDim.x + threadIdx.x;
    if (i < Bn * Vn * Hn) xh[i] += fmaxf(Ux[i] + agg[i], 0.f);
}

// ---------------- final: out = sigmoid(h @ W2 + b2), one wave per edge ----------------
__global__ __launch_bounds__(256) void k_out2(const float* __restrict__ h,
                                              const float* __restrict__ W2,
                                              const float* __restrict__ b2,
                                              float* __restrict__ out) {
    int gw = (int)((blockIdx.x * (size_t)blockDim.x + threadIdx.x) >> 5);
    int lane = threadIdx.x & 31;
    if (gw >= Bn * En) return;
    float4 hv = *(const float4*)(h + (size_t)gw * Hn + lane * 4);
    int k = lane * 4;
    float a0 = hv.x * W2[(k + 0) * 2] + hv.y * W2[(k + 1) * 2] +
               hv.z * W2[(k + 2) * 2] + hv.w * W2[(k + 3) * 2];
    float a1 = hv.x * W2[(k + 0) * 2 + 1] + hv.y * W2[(k + 1) * 2 + 1] +
               hv.z * W2[(k + 2) * 2 + 1] + hv.w * W2[(k + 3) * 2 + 1];
    for (int off = 16; off; off >>= 1) {
        a0 += __shfl_down(a0, off);
        a1 += __shfl_down(a1, off);
    }
    if (lane == 0) {
        out[(size_t)gw * 2 + 0] = sigmoidf_(a0 + b2[0]);
        out[(size_t)gw * 2 + 1] = sigmoidf_(a1 + b2[1]);
    }
}

extern "C" void kernel_launch(void* const* d_in, const int* in_sizes, int n_in,
                              void* d_out, int out_size, void* d_ws, size_t ws_size,
                              hipStream_t stream) {
    const float* x      = (const float*)d_in[0];
    const float* e      = (const float*)d_in[1];
    const int*   eidx   = (const int*)  d_in[2];
    const float* node_W = (const float*)d_in[3];
    const float* node_b = (const float*)d_in[4];
    const float* edge_W = (const float*)d_in[5];
    const float* edge_b = (const float*)d_in[6];
    const float* U_W    = (const float*)d_in[7];
    const float* U_b    = (const float*)d_in[8];
    const float* V_Wp   = (const float*)d_in[9];
    const float* V_b    = (const float*)d_in[10];
    const float* A_W    = (const float*)d_in[11];
    const float* A_b    = (const float*)d_in[12];
    const float* B_Wp   = (const float*)d_in[13];
    const float* B_b    = (const float*)d_in[14];
    const float* C_Wp   = (const float*)d_in[15];
    const float* C_b    = (const float*)d_in[16];
    const float* oW1    = (const float*)d_in[17];
    const float* ob1    = (const float*)d_in[18];
    const float* oW2    = (const float*)d_in[19];
    const float* ob2    = (const float*)d_in[20];

    float* ws = (float*)d_ws;
    const size_t NVH = (size_t)Bn * Vn * Hn;
    float* xh  = ws;
    float* Ux  = xh  + NVH;
    float* Vx  = Ux  + NVH;
    float* Bx  = Vx  + NVH;
    float* Cx  = Bx  + NVH;
    float* agg = Cx  + NVH;
    float* eh  = agg + NVH;

    k_embed_nodes<<<dim3((Bn * Vn * Hn + 255) / 256), 256, 0, stream>>>(x, node_W, node_b, xh);
    {
        size_t tot = (size_t)Bn * En * Hn;
        int blocks = (int)((tot + 256 * 8 - 1) / (256 * 8));
        k_embed_edges<<<dim3(blocks), 256, 0, stream>>>(e, edge_W, edge_b, eh);
    }

    const int HH = Hn * Hn;
    dim3 gEdge((En + TILE_M - 1) / TILE_M, Bn);
    dim3 gNode((4 * Vn * Hn + 255) / 256, Bn);
    for (int l = 0; l < Ln; ++l) {
        k_zero<<<dim3((int)((NVH + 255) / 256)), 256, 0, stream>>>(agg, (int)NVH);
        k_node<<<gNode, 256, 0, stream>>>(xh,
                                          U_W + l * HH, U_b + l * Hn,
                                          V_Wp + l * HH, V_b + l * Hn,
                                          B_Wp + l * HH, B_b + l * Hn,
                                          C_Wp + l * HH, C_b + l * Hn,
                                          Ux, Vx, Bx, Cx);
        k_edge<<<gEdge, 256, 0, stream>>>(eh, A_W + l * HH, A_b + l * Hn,
                                          Bx, Cx, Vx, agg, eidx, 0);
        k_xupd<<<dim3((int)((NVH + 255) / 256)), 256, 0, stream>>>(xh, Ux, agg);
    }
    // final edge-GEMM: h = relu(eh @ out_W1 + out_b1), overwrites eh
    k_edge<<<gEdge, 256, 0, stream>>>(eh, oW1, ob1, nullptr, nullptr, nullptr, nullptr, eidx, 1);
    // out = sigmoid(h @ out_W2 + out_b2)
    int nwaves = Bn * En;
    k_out2<<<dim3((nwaves * 32 + 255) / 256), 256, 0, stream>>>(eh, oW2, ob2, (float*)d_out);
}
